// SubNetwork_74466142978244
// MI455X (gfx1250) — matrix operand order repaired and checked by reading, not compiled, not run on hardware
//
#include <hip/hip_runtime.h>

// MI455X (gfx1250) wave32 WMMA implementation.
// Roofline: GEMM1 = 2.75e11 FLOP, all operands L2-resident (emb 8MB, W1 16MB)
// -> compute bound on v_wmma_f32_16x16x32_bf16. fp32 accuracy recovered via
// hi/lo bf16 split (3 WMMAs per fp32 product-tile). W1 tiles are streamed to
// LDS with gfx1250 async loads (ASYNCcnt) and double-buffered so the fetch
// hides behind the VALU gather/split of the A tile.

typedef __attribute__((ext_vector_type(16))) __bf16 v16bf;
typedef __attribute__((ext_vector_type(8)))  float  v8f;
typedef __attribute__((ext_vector_type(4)))  int    v4i;
typedef __attribute__((ext_vector_type(4)))  float  v4f;
typedef __attribute__((address_space(3))) unsigned short lds_us;

#define DIMB 32768
#define DIMT 2048
#define DIMD 1024
#define DIMK 4096   // S*D
#define DIMH 1024
#define DIMO 64
#define LDSTR 40    // padded LDS row stride (bf16 elems): 80B = 5*16B, conflict-free

struct Frag2 { v4i a, b; };

__device__ __forceinline__ v16bf make_frag(const unsigned short* p0,
                                           const unsigned short* p1) {
  Frag2 f;
  f.a = *(const v4i*)p0;
  f.b = *(const v4i*)p1;
  return __builtin_bit_cast(v16bf, f);
}

__device__ __forceinline__ void split2(float x, unsigned short& h, unsigned short& l) {
  unsigned u = __float_as_uint(x);
  h = (unsigned short)(u >> 16);                       // exact truncated hi
  float r = x - __uint_as_float(u & 0xffff0000u);      // exact residual
  l = (unsigned short)(__float_as_uint(r) >> 16);
}

__device__ __forceinline__ v8f vzero8() {
  v8f z = {0.f, 0.f, 0.f, 0.f, 0.f, 0.f, 0.f, 0.f};
  return z;
}

// gfx1250 async memory->LDS copy (tracked by ASYNCcnt, no VGPR round-trip)
__device__ __forceinline__ void async_b128(unsigned short* lds,
                                           const unsigned short* g) {
  unsigned loff = (unsigned)(size_t)(lds_us*)lds;  // 32-bit LDS offset
  asm volatile("global_load_async_to_lds_b128 %0, %1, off"
               :: "v"(loff), "v"(g) : "memory");
}

#define WMMA_BF16(A, B, C) \
  __builtin_amdgcn_wmma_f32_16x16x32_bf16(false, (A), false, (B), (short)0, (C), false, false)

// ---------------------------------------------------------------------------
// One-time: src[K][N] f32 -> dhi[N][K], dlo[N][K] bf16 (transpose + split)
// ---------------------------------------------------------------------------
__global__ __launch_bounds__(256) void prep_split_transpose(
    const float* __restrict__ src, unsigned short* __restrict__ dhi,
    unsigned short* __restrict__ dlo, int K, int N) {
  __shared__ unsigned packed[64][65];
  const int k0 = blockIdx.x * 64, n0 = blockIdx.y * 64;
  const int t = threadIdx.x;
#pragma unroll
  for (int i = 0; i < 16; ++i) {
    int e = i * 256 + t;
    int r = e >> 6, c = e & 63;
    float v = src[(size_t)(k0 + r) * N + (n0 + c)];
    unsigned short h, l;
    split2(v, h, l);
    packed[r][c] = (unsigned)h | ((unsigned)l << 16);
  }
  __syncthreads();
#pragma unroll
  for (int i = 0; i < 16; ++i) {
    int e = i * 256 + t;
    int nr = e >> 6, kc = e & 63;
    unsigned p = packed[kc][nr];
    size_t o = (size_t)(n0 + nr) * K + (k0 + kc);
    dhi[o] = (unsigned short)(p & 0xffffu);
    dlo[o] = (unsigned short)(p >> 16);
  }
}

// ---------------------------------------------------------------------------
// GEMM1: hidden[b][h] = (emb[R[b,s]] - emb[L[b,s]-1]) @ W1 + b1, stored split
// block = 128 (b) x 128 (h); 8 waves = 4(M) x 2(N); wave tile 32 x 64
// ---------------------------------------------------------------------------
__global__ __launch_bounds__(256) void gemm1_kernel(
    const float* __restrict__ emb, const int* __restrict__ lefts,
    const int* __restrict__ rights, const unsigned short* __restrict__ w1h,
    const unsigned short* __restrict__ w1l, const float* __restrict__ b1,
    unsigned short* __restrict__ Hhi, unsigned short* __restrict__ Hlo) {
  __shared__ unsigned short Ah[128 * LDSTR], Al[128 * LDSTR];
  __shared__ unsigned short Bh[2][128 * LDSTR], Bl[2][128 * LDSTR];
  __shared__ int idxL[128][4], idxR[128][4];

  const int t = threadIdx.x;
  const int b0 = blockIdx.x * 128;
  const int n0 = blockIdx.y * 128;

  if (t < 128) {
#pragma unroll
    for (int s = 0; s < 4; ++s) {
      idxL[t][s] = lefts[(size_t)(b0 + t) * 4 + s] - 1;
      idxR[t][s] = rights[(size_t)(b0 + t) * 4 + s];
    }
  }

  const int lane = t & 31;
  const int wave = t >> 5;
  const int wm = wave & 3;   // row group: 32*wm
  const int wn = wave >> 2;  // col group: 64*wn
  const int lh = lane >> 4;  // lane half
  const int li = lane & 15;

  v8f acc[2][4];
#pragma unroll
  for (int i = 0; i < 2; ++i)
#pragma unroll
    for (int j = 0; j < 4; ++j) acc[i][j] = vzero8();

  const int row = t >> 1;         // staging row 0..127
  const int seg = (t & 1) * 16;   // 16-elem segment within 32-wide k chunk

  // global base pointers for this thread's W1^T staging row
  const unsigned short* grh = w1h + (size_t)(n0 + row) * DIMK + seg;
  const unsigned short* grl = w1l + (size_t)(n0 + row) * DIMK + seg;

  // prologue: async-fetch W1 chunk 0 into buffer 0 (4 ops in flight)
  async_b128(&Bh[0][row * LDSTR + seg],     grh);
  async_b128(&Bh[0][row * LDSTR + seg + 8], grh + 8);
  async_b128(&Bl[0][row * LDSTR + seg],     grl);
  async_b128(&Bl[0][row * LDSTR + seg + 8], grl + 8);

  __syncthreads();  // publish idxL/idxR

  for (int kk = 0; kk < DIMK; kk += 32) {
    const int buf = (kk >> 5) & 1;
    const int s = kk >> 10;              // span index (chunks never cross D)
    const int ko = (kk & (DIMD - 1)) + seg;

    // ---- async-fetch NEXT W1 chunk into the other buffer (overlaps A work)
    if (kk + 32 < DIMK) {
      unsigned short* nbh = &Bh[buf ^ 1][row * LDSTR + seg];
      unsigned short* nbl = &Bl[buf ^ 1][row * LDSTR + seg];
      async_b128(nbh,     grh + kk + 32);
      async_b128(nbh + 8, grh + kk + 40);
      async_b128(nbl,     grl + kk + 32);
      async_b128(nbl + 8, grl + kk + 40);
    }

    // ---- stage A: gather boundary rows, diff, split, pack -> LDS
    {
      const float* er = emb + (size_t)idxR[row][s] * DIMD + ko;
      const float* el = emb + (size_t)idxL[row][s] * DIMD + ko;
      unsigned hw[8], lw[8];
#pragma unroll
      for (int j = 0; j < 4; ++j) {
        v4f r4 = *(const v4f*)(er + j * 4);
        v4f l4 = *(const v4f*)(el + j * 4);
#pragma unroll
        for (int q = 0; q < 4; ++q) {
          float d = r4[q] - l4[q];
          unsigned short h, l;
          split2(d, h, l);
          int e = j * 4 + q;
          if (e & 1) { hw[e >> 1] |= (unsigned)h << 16; lw[e >> 1] |= (unsigned)l << 16; }
          else       { hw[e >> 1]  = h;                 lw[e >> 1]  = l; }
        }
      }
      v4i h0 = {(int)hw[0], (int)hw[1], (int)hw[2], (int)hw[3]};
      v4i h1 = {(int)hw[4], (int)hw[5], (int)hw[6], (int)hw[7]};
      v4i l0 = {(int)lw[0], (int)lw[1], (int)lw[2], (int)lw[3]};
      v4i l1 = {(int)lw[4], (int)lw[5], (int)lw[6], (int)lw[7]};
      *(v4i*)(Ah + row * LDSTR + seg)     = h0;
      *(v4i*)(Ah + row * LDSTR + seg + 8) = h1;
      *(v4i*)(Al + row * LDSTR + seg)     = l0;
      *(v4i*)(Al + row * LDSTR + seg + 8) = l1;
    }

    // in-order async completion: keep only the 4 just-issued ops outstanding
    if (kk + 32 < DIMK) {
      asm volatile("s_wait_asynccnt 0x4" ::: "memory");
    } else {
      asm volatile("s_wait_asynccnt 0x0" ::: "memory");
    }
    __syncthreads();

    // ---- compute: 2x4 C tiles, hi/lo split => 3 WMMAs each
    const unsigned short* BhB = &Bh[buf][0];
    const unsigned short* BlB = &Bl[buf][0];
    v16bf afh[2], afl[2];
#pragma unroll
    for (int mt = 0; mt < 2; ++mt) {
      const int m = wm * 32 + mt * 16 + li;
      const unsigned short* ap = Ah + m * LDSTR + 16 * lh;  // K contiguous per lane
      const unsigned short* aq = Al + m * LDSTR + 16 * lh;
      afh[mt] = make_frag(ap, ap + 8);
      afl[mt] = make_frag(aq, aq + 8);
    }
#pragma unroll
    for (int nt = 0; nt < 4; ++nt) {
      const int n = wn * 64 + nt * 16 + li;
      const unsigned short* bp = BhB + n * LDSTR + 16 * lh;  // K contiguous per lane
      const unsigned short* bq = BlB + n * LDSTR + 16 * lh;
      v16bf bfh = make_frag(bp, bp + 8);
      v16bf bfl = make_frag(bq, bq + 8);
#pragma unroll
      for (int mt = 0; mt < 2; ++mt) {
        v8f c = acc[mt][nt];
        c = WMMA_BF16(afh[mt], bfl, c);
        c = WMMA_BF16(afl[mt], bfh, c);
        c = WMMA_BF16(afh[mt], bfh, c);
        acc[mt][nt] = c;
      }
    }
    __syncthreads();
  }

  // ---- epilogue: + b1, split, store hidden hi/lo
#pragma unroll
  for (int mt = 0; mt < 2; ++mt)
#pragma unroll
    for (int nt = 0; nt < 4; ++nt) {
      const int hh = n0 + wn * 64 + nt * 16 + li;
      const float bias = b1[hh];
#pragma unroll
      for (int r = 0; r < 8; ++r) {
        const int bb = b0 + wm * 32 + mt * 16 + r + 8 * lh;
        float v = acc[mt][nt][r] + bias;
        unsigned short h, l;
        split2(v, h, l);
        size_t o = (size_t)bb * DIMH + hh;
        Hhi[o] = h;
        Hlo[o] = l;
      }
    }
}

// ---------------------------------------------------------------------------
// GEMM2: scores[b][o] = hidden @ W2 + b2; block = 128 b-rows, 8 waves x 16 rows
// ---------------------------------------------------------------------------
__global__ __launch_bounds__(256) void gemm2_kernel(
    const unsigned short* __restrict__ Hhi, const unsigned short* __restrict__ Hlo,
    const unsigned short* __restrict__ w2h, const unsigned short* __restrict__ w2l,
    const float* __restrict__ b2, float* __restrict__ out) {
  __shared__ unsigned short Wh[64 * LDSTR], Wl[64 * LDSTR];
  const int t = threadIdx.x;
  const int b0 = blockIdx.x * 128;
  const int lane = t & 31, wave = t >> 5;
  const int lh = lane >> 4, li = lane & 15;

  v8f acc[4];
#pragma unroll
  for (int i = 0; i < 4; ++i) acc[i] = vzero8();

  const int row = t >> 1;        // 0..127 (only 0..63 used for staging)
  const int seg = (t & 1) * 16;

  for (int kc = 0; kc < DIMH; kc += 32) {
    if (t < 128) {
      const unsigned short* gh = w2h + (size_t)row * DIMH + kc + seg;
      const unsigned short* gl = w2l + (size_t)row * DIMH + kc + seg;
      *(v4i*)(Wh + row * LDSTR + seg)     = *(const v4i*)gh;
      *(v4i*)(Wh + row * LDSTR + seg + 8) = *(const v4i*)(gh + 8);
      *(v4i*)(Wl + row * LDSTR + seg)     = *(const v4i*)gl;
      *(v4i*)(Wl + row * LDSTR + seg + 8) = *(const v4i*)(gl + 8);
    }
    __syncthreads();

    const int bb = b0 + wave * 16 + li;
    const unsigned short* ph = Hhi + (size_t)bb * DIMH + kc + 16 * lh;
    const unsigned short* pl = Hlo + (size_t)bb * DIMH + kc + 16 * lh;
    v16bf afh = make_frag(ph, ph + 8);
    v16bf afl = make_frag(pl, pl + 8);

#pragma unroll
    for (int nt = 0; nt < 4; ++nt) {
      const unsigned short* bp = Wh + (nt * 16 + li) * LDSTR + 16 * lh;
      const unsigned short* bq = Wl + (nt * 16 + li) * LDSTR + 16 * lh;
      v16bf bfh = make_frag(bp, bp + 8);
      v16bf bfl = make_frag(bq, bq + 8);
      v8f c = acc[nt];
      c = WMMA_BF16(afh, bfl, c);
      c = WMMA_BF16(afl, bfh, c);
      c = WMMA_BF16(afh, bfh, c);
      acc[nt] = c;
    }
    __syncthreads();
  }

#pragma unroll
  for (int nt = 0; nt < 4; ++nt) {
    const int o = nt * 16 + li;
    const float bias = b2[o];
#pragma unroll
    for (int r = 0; r < 8; ++r) {
      const int bb = b0 + wave * 16 + r + 8 * lh;
      out[(size_t)bb * DIMO + o] = acc[nt][r] + bias;
    }
  }
}

// ---------------------------------------------------------------------------
extern "C" void kernel_launch(void* const* d_in, const int* in_sizes, int n_in,
                              void* d_out, int out_size, void* d_ws, size_t ws_size,
                              hipStream_t stream) {
  const float* emb    = (const float*)d_in[0];
  const int*   lefts  = (const int*)d_in[1];
  const int*   rights = (const int*)d_in[2];
  const float* W1     = (const float*)d_in[3];
  const float* b1     = (const float*)d_in[4];
  const float* W2     = (const float*)d_in[5];
  const float* b2     = (const float*)d_in[6];
  float* out = (float*)d_out;

  unsigned short* w1h = (unsigned short*)d_ws;                 // [H][K]
  unsigned short* w1l = w1h + (size_t)DIMH * DIMK;
  unsigned short* w2h = w1l + (size_t)DIMH * DIMK;             // [O][H]
  unsigned short* w2l = w2h + (size_t)DIMO * DIMH;
  unsigned short* Hh  = w2l + (size_t)DIMO * DIMH;             // [B][H]
  unsigned short* Hl  = Hh + (size_t)DIMB * DIMH;

  prep_split_transpose<<<dim3(DIMK / 64, DIMH / 64), 256, 0, stream>>>(W1, w1h, w1l, DIMK, DIMH);
  prep_split_transpose<<<dim3(DIMH / 64, DIMO / 64), 256, 0, stream>>>(W2, w2h, w2l, DIMH, DIMO);
  gemm1_kernel<<<dim3(DIMB / 128, DIMH / 128), 256, 0, stream>>>(emb, lefts, rights, w1h, w1l, b1, Hh, Hl);
  gemm2_kernel<<<dim3(DIMB / 128), 256, 0, stream>>>(Hh, Hl, w2h, w2l, b2, out);
}
